// ReasoningModule_59725815218429
// MI455X (gfx1250) — matrix operand order repaired and checked by reading, not compiled, not run on hardware
//
#include <hip/hip_runtime.h>
#include <hip/hip_bf16.h>

// Problem constants (fixed by the reference).
#define D_   2048
#define H_   16
#define HD_  128
#define KK_  8192
#define FFN_ 8192
#define MAX_STEPS_CAP 5

typedef __attribute__((ext_vector_type(16))) __bf16 v16bf;
typedef __attribute__((ext_vector_type(2)))  __bf16 v2bf;
typedef __attribute__((ext_vector_type(8)))  float  v8f;
typedef __attribute__((ext_vector_type(4)))  float  f32x4;
typedef __attribute__((ext_vector_type(2)))  unsigned int u32x2;

union FragBF { f32x4 f4[2]; v16bf v; };

#if defined(__has_builtin)
#if __has_builtin(__builtin_amdgcn_cvt_pk_bf16_f32)
#define HAVE_CVT_PK_BF16 1
#endif
#if __has_builtin(__builtin_amdgcn_perm)
#define HAVE_PERM 1
#endif
#endif

__device__ __forceinline__ unsigned short f32_to_bf16(float x) {
  unsigned int u = __float_as_uint(x);
  unsigned int r = u + 0x7FFFu + ((u >> 16) & 1u);  // round-to-nearest-even
  return (unsigned short)(r >> 16);
}
// Pack two f32 -> one dword of two bf16.
// Preferred: v_cvt_pk_bf16_f32 (1 op). Else: 2x v_add (round-half-up) +
// v_perm_b32 selecting the high halves (3 ops). Else: manual RNE (~6 ops).
__device__ __forceinline__ unsigned int pack_bf16(float lo, float hi) {
#if defined(HAVE_CVT_PK_BF16)
  v2bf r = __builtin_amdgcn_cvt_pk_bf16_f32(lo, hi);
  union { v2bf v; unsigned int u; } cvt;
  cvt.v = r;
  return cvt.u;
#elif defined(HAVE_PERM)
  unsigned int ul = __float_as_uint(lo) + 0x8000u;   // round half up
  unsigned int uh = __float_as_uint(hi) + 0x8000u;
  // bytes: dst = {uh.b3, uh.b2, ul.b3, ul.b2}
  return __builtin_amdgcn_perm(uh, ul, 0x07060302u);
#else
  return (unsigned)f32_to_bf16(lo) | ((unsigned)f32_to_bf16(hi) << 16);
#endif
}
__device__ __forceinline__ float bf16u_to_f32(unsigned int u) {
  return __uint_as_float(u << 16);
}
__device__ __forceinline__ float gelu_exact(float x) {
  return 0.5f * x * (1.0f + erff(x * 0.70710678118654752f));
}
__device__ __forceinline__ float wave_reduce(float s) {
#pragma unroll
  for (int off = 16; off > 0; off >>= 1) s += __shfl_down(s, off, 32);
  return s;
}

// ---------------------------------------------------------------------------
// GEMM: C[M,N] (bf16) = A[M,Kd] (f32) @ W[N,Kd]^T (f32), via bf16 WMMA.
// Tiles: 128x128x32, 256 threads = 8 waves laid out 4(M) x 2(N),
// each wave computes 2x4 tiles of 16x16 with v_wmma_f32_16x16x32_bf16.
// Software-pipelined: global loads for k+1 are issued before the WMMA block;
// the f32->bf16 convert + LDS store for k+1 happens after the WMMAs, into the
// other half of a double-buffered LDS tile (one barrier per iteration).
// ---------------------------------------------------------------------------
__global__ __launch_bounds__(256)
void gemm_bf16_wmma(const float* __restrict__ A, const float* __restrict__ W,
                    unsigned short* __restrict__ C, int M, int N, int Kd) {
  __shared__ __align__(16) unsigned short As[2][128 * 40];  // 128 x 32 bf16 (+8 pad)
  __shared__ __align__(16) unsigned short Bs[2][128 * 40];
  const int tid  = threadIdx.x;
  const int lane = tid & 31;
  const int wv   = tid >> 5;
  const int wm4  = wv & 3;    // wave position along M (4)
  const int wn2  = wv >> 2;   // wave position along N (2)
  const int m0   = blockIdx.y * 128;
  const int n0   = blockIdx.x * 128;
  const int l15  = lane & 15;
  const int lhi  = lane >> 4;

  v8f acc[2][4];
#pragma unroll
  for (int i = 0; i < 2; ++i)
#pragma unroll
    for (int j = 0; j < 4; ++j)
#pragma unroll
      for (int r = 0; r < 8; ++r) acc[i][j][r] = 0.0f;

  f32x4 ra[4], rb[4];  // in-flight staging registers (next k-tile)

  auto load_tiles = [&](int k0) {
#pragma unroll
    for (int i = 0; i < 4; ++i) {
      int id  = tid + i * 256;  // 0..1023
      int row = id >> 3;        // 0..127
      int c4  = id & 7;         // float4 column
      ra[i] = *reinterpret_cast<const f32x4*>(A + (size_t)(m0 + row) * Kd + k0 + c4 * 4);
      rb[i] = *reinterpret_cast<const f32x4*>(W + (size_t)(n0 + row) * Kd + k0 + c4 * 4);
    }
  };
  auto store_tiles = [&](int buf) {
#pragma unroll
    for (int i = 0; i < 4; ++i) {
      int id  = tid + i * 256;
      int row = id >> 3;
      int c4  = id & 7;
      u32x2 ap, bp;
      ap.x = pack_bf16(ra[i].x, ra[i].y);
      ap.y = pack_bf16(ra[i].z, ra[i].w);
      bp.x = pack_bf16(rb[i].x, rb[i].y);
      bp.y = pack_bf16(rb[i].z, rb[i].w);
      *reinterpret_cast<u32x2*>(&As[buf][row * 40 + c4 * 4]) = ap;
      *reinterpret_cast<u32x2*>(&Bs[buf][row * 40 + c4 * 4]) = bp;
    }
  };

  // Prologue: stage k-tile 0.
  load_tiles(0);
  store_tiles(0);
  __syncthreads();

  int cur = 0;
  for (int k0 = 0; k0 < Kd; k0 += 32) {
    const bool has_next = (k0 + 32) < Kd;
    if (has_next) load_tiles(k0 + 32);  // issue global loads early (hidden under WMMA)

    // Fragments per ISA VGPR layouts:
    // A 16x32 bf16: lane l -> M=l&15, K = {(l>>4)*8..+7, 16+(l>>4)*8..+7}
    // B 32x16 bf16: lane l -> N=l&15, K = {(l>>4)*8..+7, 16+(l>>4)*8..+7}
    FragBF afr[2];
#pragma unroll
    for (int mt = 0; mt < 2; ++mt) {
      int ar = wm4 * 32 + mt * 16 + l15;
      afr[mt].f4[0] = *reinterpret_cast<const f32x4*>(&As[cur][ar * 40 + lhi * 8]);
      afr[mt].f4[1] = *reinterpret_cast<const f32x4*>(&As[cur][ar * 40 + 16 + lhi * 8]);
    }
    FragBF bfr[4];
#pragma unroll
    for (int nt = 0; nt < 4; ++nt) {
      int bc = wn2 * 64 + nt * 16 + l15;
      bfr[nt].f4[0] = *reinterpret_cast<const f32x4*>(&Bs[cur][bc * 40 + lhi * 8]);
      bfr[nt].f4[1] = *reinterpret_cast<const f32x4*>(&Bs[cur][bc * 40 + 16 + lhi * 8]);
    }
#pragma unroll
    for (int mt = 0; mt < 2; ++mt)
#pragma unroll
      for (int nt = 0; nt < 4; ++nt)
        acc[mt][nt] = __builtin_amdgcn_wmma_f32_16x16x32_bf16(
            false, afr[mt].v, false, bfr[nt].v, (short)0, acc[mt][nt], false, false);

    if (has_next) store_tiles(cur ^ 1);  // convert + stage into the other buffer
    __syncthreads();
    cur ^= 1;
  }

  // Store C as bf16 (K/V caches are read 5x per call -> halve the bytes).
  // C/D layout: VGPR r -> M = r + 8*(lane>=16), N = lane&15.
#pragma unroll
  for (int mt = 0; mt < 2; ++mt)
#pragma unroll
    for (int nt = 0; nt < 4; ++nt)
#pragma unroll
      for (int r = 0; r < 4; ++r) {
        int m = m0 + wm4 * 32 + mt * 16 + lhi * 8 + r * 2;
        int n = n0 + wn2 * 64 + nt * 16 + l15;
        // pack two consecutive-M rows? M stride hits different rows; store singles.
        C[(size_t)m * N + n]       = f32_to_bf16(acc[mt][nt][r * 2]);
        C[(size_t)(m + 1) * N + n] = f32_to_bf16(acc[mt][nt][r * 2 + 1]);
      }
}

// ---------------------------------------------------------------------------
// f32 -> bf16 bulk convert (for FFN weights when workspace permits).
// ---------------------------------------------------------------------------
__global__ __launch_bounds__(256)
void convert_bf16_kernel(const float* __restrict__ src, unsigned short* __restrict__ dst, int n4) {
  int i = blockIdx.x * blockDim.x + threadIdx.x;
  if (i >= n4) return;
  f32x4 v = *reinterpret_cast<const f32x4*>(src + (size_t)i * 4);
  u32x2 p;
  p.x = pack_bf16(v.x, v.y);
  p.y = pack_bf16(v.z, v.w);
  *reinterpret_cast<u32x2*>(dst + (size_t)i * 4) = p;
}

// ---------------------------------------------------------------------------
// y = act(W @ x + b), W row-major [Dout, Din] (f32). One wave per row.
// ---------------------------------------------------------------------------
__global__ __launch_bounds__(256)
void matvec_kernel(const float* __restrict__ W, const float* __restrict__ b,
                   const float* __restrict__ x, float* __restrict__ y,
                   int Dout, int Din, int act, const int* __restrict__ msp, int step) {
  if (step >= *msp) return;
  int lane = threadIdx.x & 31;
  int row  = (blockIdx.x * blockDim.x + threadIdx.x) >> 5;
  if (row >= Dout) return;
  const float* w = W + (size_t)row * Din;
  float s = 0.0f;
  for (int j = lane; j < Din; j += 32) s += w[j] * x[j];
  s = wave_reduce(s);
  if (lane == 0) {
    s += b[row];
    if (act == 1) s = gelu_exact(s);
    y[row] = s;
  }
}

// bf16-weight variant (x stays f32).
__global__ __launch_bounds__(256)
void matvec_bf16_kernel(const unsigned short* __restrict__ W, const float* __restrict__ b,
                        const float* __restrict__ x, float* __restrict__ y,
                        int Dout, int Din, int act, const int* __restrict__ msp, int step) {
  if (step >= *msp) return;
  int lane = threadIdx.x & 31;
  int row  = (blockIdx.x * blockDim.x + threadIdx.x) >> 5;
  if (row >= Dout) return;
  const unsigned short* w = W + (size_t)row * Din;
  float s = 0.0f;
  for (int j = lane; j < Din; j += 32) s += bf16u_to_f32(w[j]) * x[j];
  s = wave_reduce(s);
  if (lane == 0) {
    s += b[row];
    if (act == 1) s = gelu_exact(s);
    y[row] = s;
  }
}

// ---------------------------------------------------------------------------
// scores[h, k] = (q[h,:] . Kproj[k, h*128:+128]) / sqrt(128). One wave/(k,h).
// ---------------------------------------------------------------------------
__global__ __launch_bounds__(256)
void scores_kernel(const unsigned short* __restrict__ Kp, const float* __restrict__ q,
                   float* __restrict__ sc, const int* __restrict__ msp, int step) {
  if (step >= *msp) return;
  int lane = threadIdx.x & 31;
  int gw   = (blockIdx.x * blockDim.x + threadIdx.x) >> 5;  // 0..K*H-1
  int key  = gw >> 4;
  int h    = gw & 15;
  u32x2 p = *reinterpret_cast<const u32x2*>(Kp + (size_t)key * D_ + h * HD_ + lane * 4);
  const float* qp = q + h * HD_ + lane * 4;
  float s = bf16u_to_f32(p.x & 0xFFFFu) * qp[0]
          + bf16u_to_f32(p.x >> 16)     * qp[1]
          + bf16u_to_f32(p.y & 0xFFFFu) * qp[2]
          + bf16u_to_f32(p.y >> 16)     * qp[3];
  s = wave_reduce(s);
  if (lane == 0) sc[(size_t)h * KK_ + key] = s * 0.08838834764831845f;  // 1/sqrt(128)
}

// Softmax over K=8192 per head (in place). One block per head.
__global__ __launch_bounds__(256)
void softmax_kernel(float* __restrict__ sc, const int* __restrict__ msp, int step) {
  if (step >= *msp) return;
  __shared__ float red[256];
  __shared__ float stat[2];
  float* row = sc + (size_t)blockIdx.x * KK_;
  int t = threadIdx.x;
  float m = -3.402823466e+38f;
  for (int i = t; i < KK_; i += 256) m = fmaxf(m, row[i]);
  red[t] = m; __syncthreads();
  for (int s2 = 128; s2 > 0; s2 >>= 1) { if (t < s2) red[t] = fmaxf(red[t], red[t + s2]); __syncthreads(); }
  if (t == 0) stat[0] = red[0];
  __syncthreads();
  m = stat[0];
  float sum = 0.0f;
  for (int i = t; i < KK_; i += 256) { float e = __expf(row[i] - m); row[i] = e; sum += e; }
  red[t] = sum; __syncthreads();
  for (int s2 = 128; s2 > 0; s2 >>= 1) { if (t < s2) red[t] += red[t + s2]; __syncthreads(); }
  if (t == 0) stat[1] = 1.0f / red[0];
  __syncthreads();
  float inv = stat[1];
  for (int i = t; i < KK_; i += 256) row[i] *= inv;
}

__global__ void zero_kernel(float* __restrict__ p, int n, const int* __restrict__ msp, int step) {
  if (step >= *msp) return;
  int i = blockIdx.x * blockDim.x + threadIdx.x;
  if (i < n) p[i] = 0.0f;
}

// o[h*128+d] += sum_k probs[h,k] * Vproj[k, h*128+d]; key range split over blocks.
// Block's 128 probabilities staged once into LDS (one coalesced load) instead
// of 128 redundant per-thread global loads per k iteration.
__global__ __launch_bounds__(128)
void attnv_kernel(const unsigned short* __restrict__ Vp, const float* __restrict__ probs,
                  float* __restrict__ o, const int* __restrict__ msp, int step) {
  if (step >= *msp) return;
  __shared__ float pl[128];
  int h = blockIdx.y;
  int d = threadIdx.x;             // 0..127
  int kbase = blockIdx.x * 128;
  pl[d] = probs[(size_t)h * KK_ + kbase + d];
  __syncthreads();
  float s = 0.0f;
  for (int kk = 0; kk < 128; ++kk) {
    int key = kbase + kk;
    s += pl[kk] * bf16u_to_f32((unsigned)Vp[(size_t)key * D_ + h * HD_ + d]);
  }
  atomicAdd(&o[h * HD_ + d], s);
}

// out = LayerNorm(a + r) * g + b   (single block over 2048).
__global__ __launch_bounds__(256)
void addln_kernel(const float* __restrict__ a, const float* __restrict__ r,
                  const float* __restrict__ g, const float* __restrict__ bb,
                  float* __restrict__ out, const int* __restrict__ msp, int step) {
  if (step >= *msp) return;
  __shared__ float red[256];
  __shared__ float mv[2];
  int t = threadIdx.x;
  float s = 0.0f;
  for (int i = t; i < D_; i += 256) s += a[i] + r[i];
  red[t] = s; __syncthreads();
  for (int s2 = 128; s2 > 0; s2 >>= 1) { if (t < s2) red[t] += red[t + s2]; __syncthreads(); }
  if (t == 0) mv[0] = red[0] * (1.0f / D_);
  __syncthreads();
  float m = mv[0];
  float v = 0.0f;
  for (int i = t; i < D_; i += 256) { float d = a[i] + r[i] - m; v += d * d; }
  red[t] = v; __syncthreads();
  for (int s2 = 128; s2 > 0; s2 >>= 1) { if (t < s2) red[t] += red[t + s2]; __syncthreads(); }
  if (t == 0) mv[1] = rsqrtf(red[0] * (1.0f / D_) + 1e-5f);
  __syncthreads();
  float rs = mv[1];
  for (int i = t; i < D_; i += 256) out[i] = (a[i] + r[i] - m) * rs * g[i] + bb[i];
}

// Halting head + ACT scalar bookkeeping. scal: [0]=cum [1]=active [2]=w_eff [3]=upd
__global__ __launch_bounds__(256)
void halt_kernel(const float* __restrict__ ns,
                 const float* __restrict__ w1, const float* __restrict__ b1,
                 const float* __restrict__ w2, const float* __restrict__ b2,
                 float* __restrict__ scal, const int* __restrict__ msp, int step) {
  if (step >= *msp) return;
  __shared__ float hh[32];
  int lane = threadIdx.x & 31;
  int wv = threadIdx.x >> 5;
#pragma unroll
  for (int rr = 0; rr < 4; ++rr) {
    int row = wv * 4 + rr;
    const float* w = w1 + (size_t)row * D_;
    float s = 0.0f;
    for (int j = lane; j < D_; j += 32) s += w[j] * ns[j];
    s = wave_reduce(s);
    if (lane == 0) hh[row] = gelu_exact(s + b1[row]);
  }
  __syncthreads();
  if (threadIdx.x == 0) {
    float p = b2[0];
    for (int j = 0; j < 32; ++j) p += w2[j] * hh[j];
    p = 1.0f / (1.0f + expf(-p));
    float cum = scal[0], act = scal[1];
    float ncum = cum + p;
    bool halts = (ncum >= 1.0f - 1e-6f);
    float wgt = halts ? (1.0f - cum) : p;
    scal[2] = (act != 0.0f) ? wgt : 0.0f;
    scal[3] = act;
    if (act != 0.0f) scal[0] = ncum;
    scal[1] = (act != 0.0f && !halts) ? 1.0f : 0.0f;
  }
}

__global__ void accum_kernel(const float* __restrict__ ns, float* __restrict__ acc,
                             float* __restrict__ state, const float* __restrict__ scal,
                             const int* __restrict__ msp, int step) {
  if (step >= *msp) return;
  int i = blockIdx.x * blockDim.x + threadIdx.x;
  if (i >= D_) return;
  float weff = scal[2], upd = scal[3];
  acc[i] += weff * ns[i];
  if (upd != 0.0f) state[i] = ns[i];
}

__global__ void init_kernel(const float* __restrict__ ws_in, float* __restrict__ state,
                            float* __restrict__ acc, float* __restrict__ scal) {
  int i = blockIdx.x * blockDim.x + threadIdx.x;
  if (i < D_) { state[i] = ws_in[i]; acc[i] = 0.0f; }
  if (i == 0) { scal[0] = 0.0f; scal[1] = 1.0f; scal[2] = 0.0f; scal[3] = 0.0f; }
}

__global__ void final_kernel(const float* __restrict__ acc, const float* __restrict__ state,
                             const float* __restrict__ scal, float* __restrict__ out) {
  int i = blockIdx.x * blockDim.x + threadIdx.x;
  if (i >= D_) return;
  float cum = scal[0];
  float w = (cum < 1.0f - 1e-6f) ? (1.0f - cum) : 0.0f;
  out[i] = acc[i] + w * state[i];
}

// ---------------------------------------------------------------------------
extern "C" void kernel_launch(void* const* d_in, const int* in_sizes, int n_in,
                              void* d_out, int out_size, void* d_ws, size_t ws_size,
                              hipStream_t stream) {
  (void)in_sizes; (void)n_in; (void)out_size;
  const float* workspace   = (const float*)d_in[0];
  const float* wm          = (const float*)d_in[1];
  const float* self_in_w   = (const float*)d_in[2];
  const float* self_in_b   = (const float*)d_in[3];
  const float* self_out_w  = (const float*)d_in[4];
  const float* self_out_b  = (const float*)d_in[5];
  const float* cross_in_w  = (const float*)d_in[6];
  const float* cross_in_b  = (const float*)d_in[7];
  const float* cross_out_w = (const float*)d_in[8];
  const float* cross_out_b = (const float*)d_in[9];
  const float* ffn_w1      = (const float*)d_in[10];
  const float* ffn_b1      = (const float*)d_in[11];
  const float* ffn_w2      = (const float*)d_in[12];
  const float* ffn_b2      = (const float*)d_in[13];
  const float* ln1_g       = (const float*)d_in[14];
  const float* ln1_b       = (const float*)d_in[15];
  const float* ln2_g       = (const float*)d_in[16];
  const float* ln2_b       = (const float*)d_in[17];
  const float* ln3_g       = (const float*)d_in[18];
  const float* ln3_b       = (const float*)d_in[19];
  const float* halt_w1     = (const float*)d_in[20];
  const float* halt_b1     = (const float*)d_in[21];
  const float* halt_w2     = (const float*)d_in[22];
  const float* halt_b2     = (const float*)d_in[23];
  const int*   max_steps   = (const int*)d_in[24];

  // Workspace layout (bytes):
  //   [0, 32M)    Kproj bf16 [KK_, D_]
  //   [32M, 64M)  Vproj bf16 [KK_, D_]
  //   then f32 scratch (~0.62 MB), then optional bf16 FFN weights (64 MB).
  char* wsb = (char*)d_ws;
  const size_t KV_BYTES = (size_t)KK_ * D_ * 2;
  unsigned short* Kproj = (unsigned short*)wsb;
  unsigned short* Vproj = (unsigned short*)(wsb + KV_BYTES);
  float* f = (float*)(wsb + 2 * KV_BYTES);
  float* state  = f;               // 2048
  float* acc    = f + 2048;        // 2048
  float* qb     = f + 4096;        // 2048
  float* ob     = f + 6144;        // 2048 (attention output)
  float* s1     = f + 8192;        // 2048
  float* vb     = f + 10240;       // 2048 (self-attn V)
  float* tmp    = f + 12288;       // 2048
  float* s2     = f + 14336;       // 2048
  float* ns     = f + 16384;       // 2048 (new state)
  float* hbuf   = f + 18432;       // 8192 (FFN hidden)
  float* scores = f + 26624;       // H_*KK_ = 131072
  float* scal   = f + 157696;      // 16
  size_t base_bytes = 2 * KV_BYTES + (157712) * sizeof(float);
  size_t base_al = (base_bytes + 255) & ~(size_t)255;
  const size_t FFN_BYTES = (size_t)FFN_ * D_ * 2;  // 32 MB each
  unsigned short* ffn1_bf = (unsigned short*)(wsb + base_al);
  unsigned short* ffn2_bf = (unsigned short*)(wsb + base_al + FFN_BYTES);
  bool use_bf16_ffn = ws_size >= base_al + 2 * FFN_BYTES;

  // --- One-time: K/V projections of wm_context via bf16 WMMA GEMM ---
  {
    dim3 g(D_ / 128, KK_ / 128);  // (16, 64)
    gemm_bf16_wmma<<<g, 256, 0, stream>>>(wm, cross_in_w + (size_t)D_ * D_,     Kproj, KK_, D_, D_);
    gemm_bf16_wmma<<<g, 256, 0, stream>>>(wm, cross_in_w + (size_t)2 * D_ * D_, Vproj, KK_, D_, D_);
  }
  if (use_bf16_ffn) {
    int n4 = (FFN_ * D_) / 4;  // 4,194,304
    convert_bf16_kernel<<<(n4 + 255) / 256, 256, 0, stream>>>(ffn_w1, ffn1_bf, n4);
    convert_bf16_kernel<<<(n4 + 255) / 256, 256, 0, stream>>>(ffn_w2, ffn2_bf, n4);
  }
  init_kernel<<<8, 256, 0, stream>>>(workspace, state, acc, scal);

  const int MV_D   = (D_ * 32 + 255) / 256;   // one wave per row, Dout=2048 -> 256 blocks
  const int MV_FFN = (FFN_ * 32 + 255) / 256; // Dout=8192 -> 1024 blocks

  for (int step = 0; step < MAX_STEPS_CAP; ++step) {
    // Cross-attention (K/V precomputed; only q-projection per step).
    matvec_kernel<<<MV_D, 256, 0, stream>>>(cross_in_w, cross_in_b, state, qb, D_, D_, 0, max_steps, step);
    scores_kernel<<<(KK_ * H_) / 8, 256, 0, stream>>>(Kproj, qb, scores, max_steps, step);
    softmax_kernel<<<H_, 256, 0, stream>>>(scores, max_steps, step);
    zero_kernel<<<8, 256, 0, stream>>>(ob, D_, max_steps, step);
    attnv_kernel<<<dim3(KK_ / 128, H_), 128, 0, stream>>>(Vproj, scores, ob, max_steps, step);
    matvec_kernel<<<MV_D, 256, 0, stream>>>(cross_out_w, cross_out_b, ob, tmp, D_, D_, 0, max_steps, step);
    addln_kernel<<<1, 256, 0, stream>>>(state, tmp, ln1_g, ln1_b, s1, max_steps, step);

    // Self-attention with a single KV token: softmax(1)=1 -> o = V(s1).
    matvec_kernel<<<MV_D, 256, 0, stream>>>(self_in_w + (size_t)2 * D_ * D_, self_in_b + 2 * D_,
                                            s1, vb, D_, D_, 0, max_steps, step);
    matvec_kernel<<<MV_D, 256, 0, stream>>>(self_out_w, self_out_b, vb, tmp, D_, D_, 0, max_steps, step);
    addln_kernel<<<1, 256, 0, stream>>>(s1, tmp, ln2_g, ln2_b, s2, max_steps, step);

    // FFN (dominant per-step bytes; bf16 weights when workspace allows).
    if (use_bf16_ffn) {
      matvec_bf16_kernel<<<MV_FFN, 256, 0, stream>>>(ffn1_bf, ffn_b1, s2, hbuf, FFN_, D_, 1, max_steps, step);
      matvec_bf16_kernel<<<MV_D, 256, 0, stream>>>(ffn2_bf, ffn_b2, hbuf, tmp, D_, FFN_, 0, max_steps, step);
    } else {
      matvec_kernel<<<MV_FFN, 256, 0, stream>>>(ffn_w1, ffn_b1, s2, hbuf, FFN_, D_, 1, max_steps, step);
      matvec_kernel<<<MV_D, 256, 0, stream>>>(ffn_w2, ffn_b2, hbuf, tmp, D_, FFN_, 0, max_steps, step);
    }
    addln_kernel<<<1, 256, 0, stream>>>(s2, tmp, ln3_g, ln3_b, ns, max_steps, step);

    // Halting + ACT accumulation.
    halt_kernel<<<1, 256, 0, stream>>>(ns, halt_w1, halt_b1, halt_w2, halt_b2, scal, max_steps, step);
    accum_kernel<<<8, 256, 0, stream>>>(ns, acc, state, scal, max_steps, step);
  }

  final_kernel<<<8, 256, 0, stream>>>(acc, state, scal, (float*)d_out);
}
